// TConv_MultiHeadAttention_6184752906478
// MI455X (gfx1250) — compile-verified, matrix operand order checked by reading, not run on hardware
//
#include <hip/hip_runtime.h>
#include <math.h>

// ---------------------------------------------------------------------------
// Problem constants (from reference): B=4, DIN=256, S=2048, DM=512, H=8, DK=64
// ---------------------------------------------------------------------------
#define S_LEN   2048
#define B_SZ    4
#define DIN_    256
#define DM_     512
#define H_HEADS 8
#define DK_     64
#define NROWS   (S_LEN * B_SZ)     // 8192 (s,b) rows
#define NH      (B_SZ * H_HEADS)   // 32 heads
#define PSTRIDE 2056               // padded LDS row stride (bf16 elems): 2048+8
                                   // -> row bases 4112B apart = 4-bank rotation

typedef __attribute__((ext_vector_type(16))) __bf16 v16bf;
typedef __attribute__((ext_vector_type(8)))  float  v8f;
typedef unsigned short bf16_t;   // raw bf16 bits storage

union FragBF {
    v16bf v;
    uint4 u4[2];
    unsigned int w[8];
};

__device__ __forceinline__ unsigned short bf16_bits(float f) {
    unsigned int u = __float_as_uint(f);
    unsigned int r = u + 0x7FFFu + ((u >> 16) & 1u);   // round-to-nearest-even
    return (unsigned short)(r >> 16);
}

__device__ __forceinline__ v8f wmma_bf16(const FragBF& a, const FragBF& b, v8f c) {
    // D(f32 16x16) = A(bf16 16x32) * B(bf16 32x16) + C
    return __builtin_amdgcn_wmma_f32_16x16x32_bf16(
        /*neg_a=*/false, a.v, /*neg_b=*/false, b.v,
        /*c_mod=*/(short)0, c, /*reuse_a=*/false, /*reuse_b=*/false);
}

// Load one 16x32 bf16 fragment. Source is K-contiguous with row stride `ld`.
// ISA layout: lane(l) row M=l&15; half=l>>4 selects K-offsets half*8 and 16+half*8.
__device__ __forceinline__ FragBF load_frag(const bf16_t* __restrict__ base,
                                            int row, int ld, int kb, int half) {
    FragBF f;
    const bf16_t* p = base + (size_t)row * ld + kb + half * 8;
    f.u4[0] = *(const uint4*)(p);
    f.u4[1] = *(const uint4*)(p + 16);
    return f;
}

// ---------------------------------------------------------------------------
// Kernel 0a: x[B,DIN,S] fp32 -> Xb[(s*B+b), DIN] bf16   (the xt transpose)
// ---------------------------------------------------------------------------
__global__ void k_convert_x(const float* __restrict__ x, bf16_t* __restrict__ xb) {
    int idx = blockIdx.x * 256 + threadIdx.x;         // [0, 8192*256)
    int r = idx >> 8;                                 // row = s*B + b
    int k = idx & 255;
    int s = r >> 2;
    int b = r & 3;
    xb[idx] = bf16_bits(x[((size_t)b * DIN_ + k) * S_LEN + s]);
}

// Kernel 0b: generic fp32 -> bf16 copy
__global__ void k_convert(const float* __restrict__ src, bf16_t* __restrict__ dst) {
    int idx = blockIdx.x * 256 + threadIdx.x;
    dst[idx] = bf16_bits(src[idx]);
}

// ---------------------------------------------------------------------------
// Kernel 1: projection GEMM.  out[r,m] = sum_k Xb[r,k]*W[m,k] + bias[m]
// Wave computes a 16x64 tile (4 accumulators, shared A fragment).
// mode 0/1: store head-split  dst[n=(b*H+h)][s][d]   (Q, K)
// mode 2  : store transposed  dst[n][d][s]           (V -> Vt)
// ---------------------------------------------------------------------------
__global__ void k_proj(const bf16_t* __restrict__ X, const bf16_t* __restrict__ W,
                       const float* __restrict__ bias, bf16_t* __restrict__ dst,
                       int mode) {
    int wid  = blockIdx.x * 8 + (threadIdx.x >> 5);
    int lane = threadIdx.x & 31;
    int l16  = lane & 15, half = lane >> 4;
    int row0 = (wid >> 3) * 16;        // 512 row tiles
    int col0 = (wid & 7) * 64;         // 8 col tiles of 64 -> DM=512

    v8f acc[4];
    for (int t = 0; t < 4; ++t) {
        float bv = bias[col0 + t * 16 + l16];
        for (int r = 0; r < 8; ++r) acc[t][r] = bv;
    }
    for (int kb = 0; kb < DIN_; kb += 32) {
        FragBF a = load_frag(X, row0 + l16, DIN_, kb, half);
        for (int t = 0; t < 4; ++t) {
            FragBF b = load_frag(W, col0 + t * 16 + l16, DIN_, kb, half);
            acc[t] = wmma_bf16(a, b, acc[t]);
        }
    }
    for (int t = 0; t < 4; ++t) {
        int m = col0 + t * 16 + l16;
        int h = m >> 6, d = m & 63;
        for (int r = 0; r < 8; ++r) {
            int rg = row0 + r + 8 * half;     // global (s,b) row
            int s = rg >> 2, b = rg & 3;
            int n = b * H_HEADS + h;
            unsigned short bits = bf16_bits(acc[t][r]);
            if (mode == 2) dst[((size_t)n * DK_ + d) * S_LEN + s] = bits;   // Vt
            else           dst[((size_t)n * S_LEN + s) * DK_ + d] = bits;   // Qh/Kh
        }
    }
}

// ---------------------------------------------------------------------------
// Kernel 2: FULLY FUSED attention: scores -> softmax -> P@V.
// One workgroup (8 waves) owns a 16-row x 2048-col strip for one (head,
// q-tile). Wave w computes score columns [w*256, w*256+256) in registers
// (16 accumulators, 32 WMMAs, A fragments reused). Softmax reduces row
// max/sum with half-wave shuffles + an 8x16 LDS buffer. The normalized P
// goes (a) as fp32 straight to d_out attn_w and (b) as bf16 into a padded
// LDS strip, from which each wave re-reads its own block as WMMA A-fragments
// (ds_load_b128, 4-bank row rotation => conflict-free) and accumulates a
// 16x64 partial of P@V against L2-resident Vt (32 more WMMAs). Partials are
// summed across waves in LDS and written as bf16 AO. This eliminates
// the 512 MiB attn_w re-read: total pipeline traffic ~0.55 GiB (~24us at
// 23.3 TB/s HBM).
// ---------------------------------------------------------------------------
__global__ void k_attention(const bf16_t* __restrict__ Qh,
                            const bf16_t* __restrict__ Kh,
                            const bf16_t* __restrict__ Vt,
                            float* __restrict__ attn,
                            bf16_t* __restrict__ AO) {
    __shared__ bf16_t strip[16 * PSTRIDE];   // 64.25 KiB normalized P (bf16)
    __shared__ float  part[8][16][64];       // 32 KiB   P@V partials per wave
    __shared__ float  red[8][16];            // 0.5 KiB  softmax reductions

    int wave = threadIdx.x >> 5;             // 0..7 -> 256-wide k slab
    int lane = threadIdx.x & 31;
    int l16  = lane & 15, half = lane >> 4;
    int head = blockIdx.x >> 7;              // 128 q tiles per head
    int q0 = (blockIdx.x & 127) * 16;
    int k0 = wave * 256;

    const bf16_t* Q = Qh + (size_t)head * S_LEN * DK_;
    const bf16_t* K = Kh + (size_t)head * S_LEN * DK_;

    // ---- phase 1: scores (16 x 256 per wave, in registers) ----
    FragBF a0 = load_frag(Q, q0 + l16, DK_, 0,  half);
    FragBF a1 = load_frag(Q, q0 + l16, DK_, 32, half);

    v8f acc[16];
    for (int t = 0; t < 16; ++t)
        for (int r = 0; r < 8; ++r) acc[t][r] = 0.0f;

    for (int t = 0; t < 16; ++t) {
        int kc = k0 + t * 16 + l16;
        FragBF b0 = load_frag(K, kc, DK_, 0,  half);
        FragBF b1 = load_frag(K, kc, DK_, 32, half);
        acc[t] = wmma_bf16(a0, b0, acc[t]);
        acc[t] = wmma_bf16(a1, b1, acc[t]);
    }

    // ---- row max: per-lane partial over 16 tiles, shuffle within half ----
    float pm[8];
    for (int r = 0; r < 8; ++r) {
        float v = acc[0][r];
        for (int t = 1; t < 16; ++t) v = fmaxf(v, acc[t][r]);
        for (int m = 1; m <= 8; m <<= 1) v = fmaxf(v, __shfl_xor(v, m));
        pm[r] = v;                           // row (r + 8*half), this wave's slab
    }
    if (l16 == 0)
        for (int r = 0; r < 8; ++r) red[wave][r + 8 * half] = pm[r];
    __syncthreads();
    float rmax[8];
    for (int r = 0; r < 8; ++r) {
        float v = red[0][r + 8 * half];
        for (int w = 1; w < 8; ++w) v = fmaxf(v, red[w][r + 8 * half]);
        rmax[r] = v;
    }
    __syncthreads();                         // before red[] reuse

    // ---- exp + row sum ----
    float ps[8];
    for (int r = 0; r < 8; ++r) ps[r] = 0.0f;
    for (int t = 0; t < 16; ++t)
        for (int r = 0; r < 8; ++r) {
            float e = __expf(acc[t][r] - rmax[r]);
            acc[t][r] = e;
            ps[r] += e;
        }
    for (int r = 0; r < 8; ++r)
        for (int m = 1; m <= 8; m <<= 1) ps[r] += __shfl_xor(ps[r], m);
    if (l16 == 0)
        for (int r = 0; r < 8; ++r) red[wave][r + 8 * half] = ps[r];
    __syncthreads();
    float inv[8];
    for (int r = 0; r < 8; ++r) {
        float v = red[0][r + 8 * half];
        for (int w = 1; w < 8; ++w) v += red[w][r + 8 * half];
        inv[r] = 1.0f / v;
    }

    // ---- normalize; fp32 -> attn_w (final output), bf16 -> LDS strip ----
    float* out = attn + ((size_t)head * S_LEN + q0) * S_LEN;
    for (int t = 0; t < 16; ++t)
        for (int r = 0; r < 8; ++r) {
            float p = acc[t][r] * inv[r];
            int row = r + 8 * half;
            int col = k0 + t * 16 + l16;
            out[(size_t)row * S_LEN + col] = p;
            strip[row * PSTRIDE + col] = bf16_bits(p);
        }
    __syncthreads();

    // ---- phase 2: P@V partial over this wave's 256-wide k slab ----
    const bf16_t* V = Vt + (size_t)head * DK_ * S_LEN;
    v8f acc2[4];
    for (int t = 0; t < 4; ++t)
        for (int r = 0; r < 8; ++r) acc2[t][r] = 0.0f;

    for (int kb = k0; kb < k0 + 256; kb += 32) {
        FragBF a;                            // A from LDS (transposed layout fix)
        const bf16_t* p = strip + l16 * PSTRIDE + kb + half * 8;
        a.u4[0] = *(const uint4*)(p);
        a.u4[1] = *(const uint4*)(p + 16);
        for (int t = 0; t < 4; ++t) {
            FragBF b = load_frag(V, t * 16 + l16, S_LEN, kb, half);
            acc2[t] = wmma_bf16(a, b, acc2[t]);
        }
    }
    for (int t = 0; t < 4; ++t)
        for (int r = 0; r < 8; ++r)
            part[wave][r + 8 * half][t * 16 + l16] = acc2[t][r];
    __syncthreads();

    // ---- cross-wave reduce -> AO (flat [q*DM + n*DK + d] order) ----
    for (int j = 0; j < 4; ++j) {
        int idx = threadIdx.x + j * 256;     // 1024 outputs: 16 rows x 64 d
        int row = idx >> 6, d = idx & 63;
        float s = part[0][row][d];
        for (int w = 1; w < 8; ++w) s += part[w][row][d];
        AO[(size_t)(q0 + row) * DM_ + head * DK_ + d] = bf16_bits(s);
    }
}

// ---------------------------------------------------------------------------
// Kernel 3: Z[b,i,s] = AO2[b,s,:] . Wo[i,:] + bo[i]   (K = 512), transposed store
// AO2 is the flat AO buffer reinterpreted as [B, S, DM] (row r = b*S+s).
// ---------------------------------------------------------------------------
__global__ void k_out(const bf16_t* __restrict__ AO, const bf16_t* __restrict__ Wob,
                      const float* __restrict__ bo, float* __restrict__ Z) {
    int wid  = blockIdx.x * 8 + (threadIdx.x >> 5);
    int lane = threadIdx.x & 31;
    int l16  = lane & 15, half = lane >> 4;
    int row0 = (wid >> 2) * 16;        // 512 row tiles
    int col0 = (wid & 3) * 64;         // 4 col tiles of 64 -> DIN=256

    v8f acc[4];
    for (int t = 0; t < 4; ++t) {
        float bv = bo[col0 + t * 16 + l16];
        for (int r = 0; r < 8; ++r) acc[t][r] = bv;
    }
    for (int kb = 0; kb < DM_; kb += 32) {
        FragBF a = load_frag(AO, row0 + l16, DM_, kb, half);
        for (int t = 0; t < 4; ++t) {
            FragBF b = load_frag(Wob, col0 + t * 16 + l16, DM_, kb, half);
            acc[t] = wmma_bf16(a, b, acc[t]);
        }
    }
    for (int t = 0; t < 4; ++t) {
        int i = col0 + t * 16 + l16;
        for (int r = 0; r < 8; ++r) {
            int rg = row0 + r + 8 * half;
            int b = rg >> 11, s = rg & 2047;
            Z[((size_t)b * DIN_ + i) * S_LEN + s] = acc[t][r];
        }
    }
}

// ---------------------------------------------------------------------------
extern "C" void kernel_launch(void* const* d_in, const int* in_sizes, int n_in,
                              void* d_out, int out_size, void* d_ws, size_t ws_size,
                              hipStream_t stream) {
    const float* x  = (const float*)d_in[0];
    const float* Wq = (const float*)d_in[1];
    const float* bq = (const float*)d_in[2];
    const float* Wk = (const float*)d_in[3];
    const float* bk = (const float*)d_in[4];
    const float* Wv = (const float*)d_in[5];
    const float* bv = (const float*)d_in[6];
    const float* Wo = (const float*)d_in[7];
    const float* bo = (const float*)d_in[8];

    float* Zout = (float*)d_out;                               // [B, DIN, S]
    float* attn = Zout + (size_t)B_SZ * DIN_ * S_LEN;          // [B*H, S, S]

    // Workspace carve-up (~39 MB, all 16B aligned)
    char* ws = (char*)d_ws;
    size_t off = 0;
    bf16_t* Xb  = (bf16_t*)(ws + off); off += (size_t)NROWS * DIN_ * 2;   // 4 MiB
    bf16_t* Wqb = (bf16_t*)(ws + off); off += (size_t)DM_ * DIN_ * 2;
    bf16_t* Wkb = (bf16_t*)(ws + off); off += (size_t)DM_ * DIN_ * 2;
    bf16_t* Wvb = (bf16_t*)(ws + off); off += (size_t)DM_ * DIN_ * 2;
    bf16_t* Wob = (bf16_t*)(ws + off); off += (size_t)DIN_ * DM_ * 2;
    bf16_t* Qh  = (bf16_t*)(ws + off); off += (size_t)NH * S_LEN * DK_ * 2; // 8 MiB
    bf16_t* Kh  = (bf16_t*)(ws + off); off += (size_t)NH * S_LEN * DK_ * 2;
    bf16_t* Vt  = (bf16_t*)(ws + off); off += (size_t)NH * DK_ * S_LEN * 2;
    bf16_t* AO  = (bf16_t*)(ws + off); off += (size_t)NROWS * DM_ * 2;      // 8 MiB

    // 0) fp32 -> bf16 conversions
    k_convert_x<<<(NROWS * DIN_) / 256, 256, 0, stream>>>(x, Xb);
    k_convert<<<(DM_ * DIN_) / 256, 256, 0, stream>>>(Wq, Wqb);
    k_convert<<<(DM_ * DIN_) / 256, 256, 0, stream>>>(Wk, Wkb);
    k_convert<<<(DM_ * DIN_) / 256, 256, 0, stream>>>(Wv, Wvb);
    k_convert<<<(DIN_ * DM_) / 256, 256, 0, stream>>>(Wo, Wob);

    // 1) Q/K/V projections (512 row tiles * 8 col tiles = 4096 waves = 512 blocks)
    k_proj<<<512, 256, 0, stream>>>(Xb, Wqb, bq, Qh, 0);
    k_proj<<<512, 256, 0, stream>>>(Xb, Wkb, bk, Kh, 1);
    k_proj<<<512, 256, 0, stream>>>(Xb, Wvb, bv, Vt, 2);

    // 2) fused scores + softmax + P@V
    //    (32 heads * 128 q tiles = 4096 workgroups of 8 waves)
    k_attention<<<4096, 256, 0, stream>>>(Qh, Kh, Vt, attn, AO);

    // 3) output projection + transposed store (2048 waves = 256 blocks)
    k_out<<<256, 256, 0, stream>>>(AO, Wob, bo, Zout);
}
